// FRAP_67800353735379
// MI455X (gfx1250) — matrix operand order repaired and checked by reading, not compiled, not run on hardware
//
#include <hip/hip_runtime.h>

// ---------------------------------------------------------------------------
// FRAP fused pipeline for MI455X (gfx1250, wave32, WMMA 16x16x32 f16)
//
// Flat row index m in [0, 56*B):  source pair p = m / B, sample b = m % B,
// yc position = m % 56, output bucket = m / 7.  Each wave processes a
// 112-row job (LCM(16,7)) = 7 WMMA tiles -> all output groups wave-local.
// Since job*112 % 56 == 0, per-tile yc base positions are the constants
// {0,16,32,48,8,24,40} and pos = T[tile]+row <= 63 (table padded to 64).
//
// Inter-stage z staging uses a K-permuted layout (slot k' = 2c+sel holds
// logical channel sel?16+c:c) so each lane stores one packed f16x2 per row;
// w_comb's B matrix is packed with the same K permutation (legal since WMMA
// sums over K).
// ---------------------------------------------------------------------------

typedef __attribute__((ext_vector_type(16))) _Float16 v16h;
typedef __attribute__((ext_vector_type(2)))  _Float16 v2h;
typedef __attribute__((ext_vector_type(8)))  float    v8f;

union V16H { v16h v; _Float16 h[16]; uint4 q[2]; };

__constant__ int c_PL[8][2] = {{1,3},{5,7},{0,2},{4,6},{0,1},{2,3},{4,5},{6,7}};

__device__ __forceinline__ float sigm(float x) { return 1.0f / (1.0f + __expf(-x)); }
__device__ __forceinline__ float frelu(float x) { return x > 0.0f ? x : 0.0f; }

// ---------------------------------------------------------------------------
// Stage 1: pressure[s][i][0:16] (f16), one thread per (sample, phase).
// ---------------------------------------------------------------------------
__global__ void __launch_bounds__(256)
k_pressure(const float* __restrict__ feat,
           const float* __restrict__ emb_phase,
           const float* __restrict__ w_veh, const float* __restrict__ b_veh,
           const float* __restrict__ w_line, const float* __restrict__ b_line,
           _Float16* __restrict__ pressure, int nTot)
{
  int t = blockIdx.x * 256 + threadIdx.x;
  if (t >= nTot) return;
  int s = t >> 3;
  int i = t & 7;

  float xo[16];
  #pragma unroll
  for (int o = 0; o < 16; ++o) xo[o] = 0.0f;

  #pragma unroll
  for (int q = 0; q < 2; ++q) {
    int l   = c_PL[i][q];
    float v = feat[s * 16 + 8 + l];
    int   e = (int)feat[s * 16 + l];
    float line[8];
    #pragma unroll
    for (int k = 0; k < 4; ++k) line[k]     = sigm(v * w_veh[k] + b_veh[k]);
    #pragma unroll
    for (int k = 0; k < 4; ++k) line[4 + k] = sigm(emb_phase[e * 4 + k]);
    #pragma unroll
    for (int o = 0; o < 16; ++o) {
      float acc = b_line[o];
      #pragma unroll
      for (int k = 0; k < 8; ++k) acc += w_line[o * 8 + k] * line[k];
      xo[o] += frelu(acc);
    }
  }
  #pragma unroll
  for (int o = 0; o < 16; ++o)
    pressure[(size_t)t * 16 + o] = (_Float16)xo[o];
}

// ---------------------------------------------------------------------------
// Stage 2: padded/interleaved yc table:
//   ycpad[pos][col*2+0] = yc(pos%56, ch=col)        (col 0..15)
//   ycpad[pos][col*2+1] = yc(pos%56, ch=16+col)     (0 for ch >= 20)
// pos in [0,64): entries 56..63 duplicate 0..7 (wrap-free indexing).
// ---------------------------------------------------------------------------
__global__ void __launch_bounds__(256)
k_yc(const float* __restrict__ emb_const, const float* __restrict__ w_const,
     const float* __restrict__ b_const, const int* __restrict__ cmask,
     float* __restrict__ ycpad)
{
  int t = blockIdx.x * 256 + threadIdx.x;
  if (t >= 64 * 32) return;
  int pos  = t >> 5;
  int c2   = t & 31;
  int col  = c2 >> 1;
  int o    = (c2 & 1) ? (16 + col) : col;
  int posm = (pos >= 56) ? pos - 56 : pos;
  float v = 0.0f;
  if (o < 20) {
    int msk = cmask[posm];
    float acc = b_const[o];
    #pragma unroll
    for (int c = 0; c < 4; ++c) acc += w_const[o * 4 + c] * emb_const[msk * 4 + c];
    v = frelu(acc);
  }
  ycpad[t] = v;
}

// ---------------------------------------------------------------------------
// Stage 3: main WMMA chain.  One 112-row job per wave (7 tiles of 16 rows).
// ---------------------------------------------------------------------------
template <bool FAST>  // FAST: nB == 16384 -> shift/mask instead of division
__global__ void __launch_bounds__(256)
k_main(const _Float16* __restrict__ pressure,
       const float* __restrict__ yc_ws,
       const float* __restrict__ w_feat, const float* __restrict__ b_feat,
       const float* __restrict__ w_comb, const float* __restrict__ b_comb,
       const float* __restrict__ w_final, const float* __restrict__ b_final,
       float* __restrict__ out, int nB, int njobs)
{
  __shared__ alignas(16) float    yc_lds[64 * 32];   // 8 KB
  __shared__ alignas(16) _Float16 zbuf[8][16 * 40];  // per-wave z staging (stride 40 f16)
  __shared__ alignas(16) float    s_tile[8][112];    // per-wave per-row scalars

  const int tid = threadIdx.x;
  {
    float4*       dst = reinterpret_cast<float4*>(yc_lds);
    const float4* src = reinterpret_cast<const float4*>(yc_ws);
    #pragma unroll
    for (int i = 0; i < 2; ++i) dst[tid + 256 * i] = src[tid + 256 * i];
  }
  __syncthreads();

  const int wave = tid >> 5;
  const int lane = tid & 31;
  const int col  = lane & 15;   // WMMA column / D-layout N
  const int half = lane >> 4;   // lane half selects K sub-range
  const int job  = blockIdx.x * 8 + wave;
  if (job >= njobs) return;

  // ---- Pack B matrices into registers (f16 B layout: lane=col, half picks K16).
  // w_comb is packed with the permuted K ordering ch(k') = k'&1 ? 16+(k'>>1) : k'>>1
  // to match the interleaved z staging layout.
  V16H bfa, bfb, bca, bcb;
  #pragma unroll
  for (int kk = 0; kk < 16; ++kk) {
    int K  = (kk < 8 ? 0 : 8) + half * 8 + kk;     // physical K slot
    int ch = (K & 1) ? (16 + (K >> 1)) : (K >> 1); // logical channel for WMMA2
    bfa.h[kk] = (_Float16)w_feat[col * 32 + K];                                      // ch 0..15
    bfb.h[kk] = (col < 4) ? (_Float16)w_feat[(16 + col) * 32 + K] : (_Float16)0.0f;  // ch 16..19
    bca.h[kk] = (ch < 20) ? (_Float16)w_comb[col * 20 + ch] : (_Float16)0.0f;
    bcb.h[kk] = (col < 4 && ch < 20) ? (_Float16)w_comb[(16 + col) * 20 + ch] : (_Float16)0.0f;
  }
  const float biasFa = b_feat[col];
  const float biasFb = (col < 4) ? b_feat[16 + col] : 0.0f;
  const float biasCa = b_comb[col];
  const float biasCb = (col < 4) ? b_comb[16 + col] : 0.0f;
  const float wfa    = w_final[col];
  const float wfb    = (col < 4) ? w_final[16 + col] : 0.0f;
  const float bfin   = b_final[0];

  _Float16* zr = zbuf[wave];
  float*    st = s_tile[wave];

  // Per-lane invariant LDS bases; all per-tile/row deltas are immediates.
  const float*  ycb  = yc_lds + half * 256 + col * 2;                                // + (T+r)*32
  _Float16*     zst  = zr + half * 320 + col * 2;                                    // + r*40
  const uint4*  zldA = reinterpret_cast<const uint4*>(zr + col * 40 + half * 8);
  const uint4*  zldB = reinterpret_cast<const uint4*>(zr + col * 40 + 16 + half * 8);
  float*        stb  = st + half * 8;                                                // + tile*16

  const int jobu  = __builtin_amdgcn_readfirstlane(job);
  const int mbase = jobu * 112;

  const int T56[7] = {0, 16, 32, 48, 8, 24, 40};  // (tile*16) % 56, compile-time

  #pragma unroll
  for (int tile = 0; tile < 7; ++tile) {
    const int m0 = mbase + tile * 16;  // scalar
    int p, b0;
    if (FAST) { p = m0 >> 14; b0 = m0 & 16383; }
    else      { p = m0 / nB;  b0 = m0 - p * nB; }
    const int pi  = p / 7;                         // scalar (SALU)
    const int c7  = p - pi * 7;
    const int pj  = c7 + (c7 >= pi ? 1 : 0);
    const int smp = b0 + col;                      // lane's sample

    // A tile (16x32 f16): K0-15 = pressure[smp][pi][:], K16-31 = pressure[smp][pj][:]
    const _Float16* pr = pressure + (size_t)smp * 128 + half * 8;  // varying part
    V16H a;
    a.q[0] = *reinterpret_cast<const uint4*>(pr + pi * 16);
    a.q[1] = *reinterpret_cast<const uint4*>(pr + pj * 16);

    v8f xfa = {}, xfb = {};
    xfa = __builtin_amdgcn_wmma_f32_16x16x32_f16(false, a.v, false, bfa.v, (short)0, xfa, false, false);
    xfb = __builtin_amdgcn_wmma_f32_16x16x32_f16(false, a.v, false, bfb.v, (short)0, xfb, false, false);

    // bias + relu + yc scale (one ds_load_b64 per row, immediate offsets),
    // store one packed f16x2 per row into the K-permuted z layout.
    #pragma unroll
    for (int r = 0; r < 8; ++r) {
      const float2 y = *reinterpret_cast<const float2*>(ycb + (T56[tile] + r) * 32);
      float za = frelu(xfa[r] + biasFa) * y.x;
      float zb = frelu(xfb[r] + biasFb) * y.y;   // y.y == 0 for ch >= 20
      v2h zpk;
      zpk.x = (_Float16)za;
      zpk.y = (_Float16)zb;
      *reinterpret_cast<v2h*>(zst + r * 40) = zpk;  // single ds_store_b32
    }
    __asm__ volatile("s_wait_dscnt 0" ::: "memory");

    // reload z in WMMA A layout (two ds_load_b128)
    V16H a2;
    a2.q[0] = *zldA;
    a2.q[1] = *zldB;

    v8f ga = {}, gb = {};
    ga = __builtin_amdgcn_wmma_f32_16x16x32_f16(false, a2.v, false, bca.v, (short)0, ga, false, false);
    gb = __builtin_amdgcn_wmma_f32_16x16x32_f16(false, a2.v, false, bcb.v, (short)0, gb, false, false);

    // final 1x20 dot per row via shfl_xor butterfly over each 16-lane half
    float sv[8];
    #pragma unroll
    for (int r = 0; r < 8; ++r) {
      float part = frelu(ga[r] + biasCa) * wfa + frelu(gb[r] + biasCb) * wfb;
      part += __shfl_xor(part, 1, 32);
      part += __shfl_xor(part, 2, 32);
      part += __shfl_xor(part, 4, 32);
      part += __shfl_xor(part, 8, 32);
      sv[r] = frelu(part + bfin);
    }
    if ((lane & 15) == 0) {  // one exec toggle per tile; two ds_store_b128
      float4 s0 = make_float4(sv[0], sv[1], sv[2], sv[3]);
      float4 s1 = make_float4(sv[4], sv[5], sv[6], sv[7]);
      *reinterpret_cast<float4*>(stb + tile * 16)     = s0;
      *reinterpret_cast<float4*>(stb + tile * 16 + 4) = s1;
    }
  }
  __asm__ volatile("s_wait_dscnt 0" ::: "memory");

  // 112 rows -> 16 outputs (groups of 7), all wave-local. Lanes 0..15 write.
  if (lane < 16) {
    float o = 0.0f;
    #pragma unroll
    for (int j = 0; j < 7; ++j) o += st[lane * 7 + j];
    out[jobu * 16 + lane] = o;
  }
}

// ---------------------------------------------------------------------------
extern "C" void kernel_launch(void* const* d_in, const int* in_sizes, int n_in,
                              void* d_out, int out_size, void* d_ws, size_t ws_size,
                              hipStream_t stream)
{
  const float* feat      = (const float*)d_in[0];
  const float* emb_phase = (const float*)d_in[1];
  const float* w_veh     = (const float*)d_in[2];
  const float* b_veh     = (const float*)d_in[3];
  const float* w_line    = (const float*)d_in[4];
  const float* b_line    = (const float*)d_in[5];
  const float* emb_const = (const float*)d_in[6];
  const float* w_feat    = (const float*)d_in[7];
  const float* b_feat    = (const float*)d_in[8];
  const float* w_const   = (const float*)d_in[9];
  const float* b_const   = (const float*)d_in[10];
  const float* w_comb    = (const float*)d_in[11];
  const float* b_comb    = (const float*)d_in[12];
  const float* w_final   = (const float*)d_in[13];
  const float* b_final   = (const float*)d_in[14];
  const int*   cmask     = (const int*)d_in[15];
  float* out = (float*)d_out;

  const int nB = in_sizes[0] / 16;                 // 16384

  _Float16* pressure = (_Float16*)d_ws;            // nB*8*16 f16 = 4 MB
  float* yc = (float*)((char*)d_ws + (size_t)nB * 8 * 16 * sizeof(_Float16));  // 64*32 f32

  {
    int nTot = nB * 8;
    k_pressure<<<(nTot + 255) / 256, 256, 0, stream>>>(
        feat, emb_phase, w_veh, b_veh, w_line, b_line, pressure, nTot);
  }
  k_yc<<<(64 * 32 + 255) / 256, 256, 0, stream>>>(emb_const, w_const, b_const, cmask, yc);
  {
    int njobs  = (nB * 56) / 112;                  // = nB/2 = 8192
    int blocks = (njobs + 7) / 8;                  // 8 waves (jobs) per 256-thread block
    if (nB == 16384) {
      k_main<true><<<blocks, 256, 0, stream>>>(pressure, yc, w_feat, b_feat,
                                               w_comb, b_comb, w_final, b_final,
                                               out, nB, njobs);
    } else {
      k_main<false><<<blocks, 256, 0, stream>>>(pressure, yc, w_feat, b_feat,
                                                w_comb, b_comb, w_final, b_final,
                                                out, nB, njobs);
    }
  }
}
